// GML_2138893713780
// MI455X (gfx1250) — compile-verified, matrix operand order checked by reading, not run on hardware
//
#include <hip/hip_runtime.h>
#include <hip/hip_bf16.h>

#define N_NODES   200000
#define F_DIM     256
#define M_DIM     128
#define K_MIX     8
#define ATT_DIM   128
#define N_GRAPHS  1024

typedef __bf16 bf16_t;
typedef bf16_t v16bf __attribute__((ext_vector_type(16)));
typedef float  v8f   __attribute__((ext_vector_type(8)));

union Frag {
    v16bf v;
    unsigned short s[16];
    uint4 q[2];
};

__device__ __forceinline__ unsigned short f2bf(float f) {
    unsigned int u = __float_as_uint(f);
    unsigned int r = u + 0x7FFFu + ((u >> 16) & 1u);   // round-to-nearest-even
    return (unsigned short)(r >> 16);
}

// ---------------------------------------------------------------------------
// Prep: gT[j][f] = bf16(g[f][j])  (1024 x 256),  w1b = bf16(w1) row-major
// ---------------------------------------------------------------------------
__global__ __launch_bounds__(256) void prep_kernel(
    const float* __restrict__ g, const float* __restrict__ w1,
    unsigned short* __restrict__ gT, unsigned short* __restrict__ w1b)
{
    int idx = blockIdx.x * 256 + threadIdx.x;
    if (idx < 1024 * 256) {
        int j = idx >> 8, f = idx & 255;
        gT[idx] = f2bf(g[f * 1024 + j]);
    } else {
        int i = idx - 1024 * 256;
        if (i < 128 * 128) w1b[i] = f2bf(w1[i]);
    }
}

// ---------------------------------------------------------------------------
// Fused per-node kernel. 8 waves/block, 16 nodes/wave.
// Pipeline: half-outer / mixture-inner (st = half*8 + k). Each half's 4
// m-tiles accumulate over 8 consecutive stages, then retire to global H,
// so only 4 H-accumulator tiles (32 VGPRs) are live at a time.
// Double-buffered LDS fed by GLOBAL_LOAD_ASYNC_TO_LDS_B128 (ASYNCcnt).
// ---------------------------------------------------------------------------
__global__ __launch_bounds__(256, 1) void fused_node_kernel(
    const float* __restrict__ x,
    const unsigned char* __restrict__ gT,    // bf16 [1024][256]
    const float* __restrict__ mu,            // [8][128]
    const float* __restrict__ sigma,         // [8][128]
    const unsigned char* __restrict__ w1b,   // bf16 [128][128]
    const float* __restrict__ w2,            // [128]
    float* __restrict__ Hout,                // [N][128]
    float* __restrict__ scores)              // [N]
{
    __shared__ __align__(16) unsigned char smem[65536];  // 2 x 32KB ping-pong

    const int tid  = threadIdx.x;
    const int wv   = tid >> 5;
    const int lane = tid & 31;
    const int li   = lane & 15;
    const int h    = lane >> 4;
    const int nb   = blockIdx.x * 128 + wv * 16;
    const bool active = (nb < N_NODES);
    const unsigned lds_base = (unsigned)(uintptr_t)smem;  // LDS aperture: low 32 bits

    // ---- prologue: kick off async DMA of stage 0 (half 0, mixture 0) ----
    {
        const unsigned char* src = gT;            // stage 0 source
        for (int i = tid; i < 2048; i += 256) {
            unsigned long long ga = (unsigned long long)(uintptr_t)(src + i * 16);
            unsigned la = lds_base + (unsigned)(i * 16);
            asm volatile("global_load_async_to_lds_b128 %0, %1, off"
                         :: "v"(la), "v"(ga) : "memory");
        }
    }

    // ---- A fragments of x for this wave's 16 rows, all 8 K-steps (F=256) ----
    Frag afr[8];
    if (active) {
        const float* xr = x + (size_t)(nb + li) * F_DIM;
#pragma unroll
        for (int s = 0; s < 8; ++s) {
            int f0 = 32 * s + 8 * h;
            float4 c0 = *(const float4*)(xr + f0);
            float4 c1 = *(const float4*)(xr + f0 + 4);
            float4 c2 = *(const float4*)(xr + f0 + 16);
            float4 c3 = *(const float4*)(xr + f0 + 20);
            float tf[16] = { c0.x, c0.y, c0.z, c0.w, c1.x, c1.y, c1.z, c1.w,
                             c2.x, c2.y, c2.z, c2.w, c3.x, c3.y, c3.z, c3.w };
#pragma unroll
            for (int i = 0; i < 16; ++i) afr[s].s[i] = f2bf(tf[i]);
        }
    }

    // ---- pipeline: half outer, mixture inner; st = half*8 + k ----
    for (int half = 0; half < 2; ++half) {
        v8f Hacc4[4];   // accumulators for this half's 4 m-tiles
#pragma unroll
        for (int tt = 0; tt < 4; ++tt)
            Hacc4[tt] = v8f{0.f,0.f,0.f,0.f,0.f,0.f,0.f,0.f};

        for (int k = 0; k < 8; ++k) {
            const int st = half * 8 + k;
            const int p  = st & 1;               // ping-pong buffer parity

            // my stage-st async loads (issued last iteration) have landed
            asm volatile("s_wait_asynccnt 0x0" ::: "memory");
            // everyone's stage-st loads done; stage-(st-1) reads done
            __syncthreads();

            // issue DMA for stage st+1 into the other buffer (overlaps compute)
            if (st + 1 < 16) {
                const int h1 = (st + 1) >> 3, k1 = (st + 1) & 7, p1 = (st + 1) & 1;
                const unsigned char* src = gT + (size_t)k1 * 65536 + (size_t)h1 * 32768;
                unsigned ldsb = lds_base + (unsigned)(p1 * 32768);
                for (int i = tid; i < 2048; i += 256) {
                    unsigned long long ga = (unsigned long long)(uintptr_t)(src + i * 16);
                    unsigned la = ldsb + (unsigned)(i * 16);
                    asm volatile("global_load_async_to_lds_b128 %0, %1, off"
                                 :: "v"(la), "v"(ga) : "memory");
                }
            }

            if (active) {
                const unsigned char* buf = smem + p * 32768;  // 64 cols x 512B
                v8f acc[4];
#pragma unroll
                for (int tt = 0; tt < 4; ++tt)
                    acc[tt] = v8f{0.f,0.f,0.f,0.f,0.f,0.f,0.f,0.f};
                // K-step outer: one A fragment feeds 4 back-to-back WMMAs.
#pragma unroll
                for (int s = 0; s < 8; ++s) {
#pragma unroll
                    for (int tt = 0; tt < 4; ++tt) {
                        Frag bfr;
                        int off = (tt * 16 + li) * 512 + 64 * s + 16 * h;
                        bfr.q[0] = *(const uint4*)(buf + off);
                        bfr.q[1] = *(const uint4*)(buf + off + 32);
                        acc[tt] = __builtin_amdgcn_wmma_f32_16x16x32_bf16(
                            false, afr[s].v, false, bfr.v, (short)0, acc[tt],
                            false, false);
                    }
                    __builtin_amdgcn_sched_barrier(0);
                }
#pragma unroll
                for (int tt = 0; tt < 4; ++tt) {
                    const int mcol = (half * 4 + tt) * 16 + li;  // m index
                    float muv = mu[k * M_DIM + mcol];
                    float sg  = sigma[k * M_DIM + mcol];
                    float inv = -0.5f / (1e-15f + sg * sg);
#pragma unroll
                    for (int v = 0; v < 8; ++v) {
                        float o = acc[tt][v];
                        float d = o - muv;
                        Hacc4[tt][v] += o * __expf(inv * d * d);
                    }
                }
            }
        }

        if (active) {   // this half's 4 m-tiles are complete: retire them
            float* Hr = Hout + (size_t)nb * M_DIM;
#pragma unroll
            for (int tt = 0; tt < 4; ++tt)
#pragma unroll
                for (int v = 0; v < 8; ++v)
                    Hr[(v + 8 * h) * M_DIM + (half * 4 + tt) * 16 + li] =
                        Hacc4[tt][v];
        }
    }

    // ---- phase 2: attention GEMM via LDS round trip ----
    __syncthreads();
    {   // stage w1 bf16 into smem[32768..65536)
        const uint4* src = (const uint4*)w1b;
        uint4* dst = (uint4*)(smem + 32768);
        for (int i = tid; i < 2048; i += 256) dst[i] = src[i];
    }
    if (active) {  // re-read own H tile (L2-hot), stage as bf16 (16x128 = 4KB)
        const float* Hr = Hout + (size_t)nb * M_DIM;
        unsigned short* hw = (unsigned short*)(smem + wv * 4096);
#pragma unroll
        for (int t = 0; t < 8; ++t)
#pragma unroll
            for (int v = 0; v < 8; ++v) {
                int idx = (v + 8 * h) * M_DIM + t * 16 + li;
                hw[idx] = f2bf(Hr[idx]);
            }
    }
    __syncthreads();

    if (active) {
        Frag ha[4];
#pragma unroll
        for (int s = 0; s < 4; ++s) {
            int off = wv * 4096 + li * 256 + 64 * s + 16 * h;
            ha[s].q[0] = *(const uint4*)(smem + off);
            ha[s].q[1] = *(const uint4*)(smem + off + 32);
        }
        float part[8] = {0.f,0.f,0.f,0.f,0.f,0.f,0.f,0.f};
#pragma unroll
        for (int t = 0; t < 8; ++t) {
            const int tcol = t * 16 + li;              // attention unit index
            v8f acc = v8f{0.f,0.f,0.f,0.f,0.f,0.f,0.f,0.f};
#pragma unroll
            for (int s = 0; s < 4; ++s) {
                Frag wb;
                int off = 32768 + tcol * 256 + 64 * s + 16 * h;
                wb.q[0] = *(const uint4*)(smem + off);
                wb.q[1] = *(const uint4*)(smem + off + 32);
                acc = __builtin_amdgcn_wmma_f32_16x16x32_bf16(
                    false, ha[s].v, false, wb.v, (short)0, acc, false, false);
            }
            float w2v = w2[tcol];
#pragma unroll
            for (int v = 0; v < 8; ++v)
                part[v] += tanhf(acc[v]) * w2v;
        }
        // reduce over the 16 lanes sharing a row (xor masks stay in-half)
#pragma unroll
        for (int mask = 1; mask < 16; mask <<= 1)
#pragma unroll
            for (int v = 0; v < 8; ++v)
                part[v] += __shfl_xor(part[v], mask, 32);
        if (li == 0) {
#pragma unroll
            for (int v = 0; v < 8; ++v)
                scores[nb + v + 8 * h] = part[v];
        }
    }
}

// ---------------------------------------------------------------------------
// Per-graph segment softmax + weighted pooling (batch is sorted).
// ---------------------------------------------------------------------------
__global__ __launch_bounds__(256) void pool_kernel(
    const float* __restrict__ H, const float* __restrict__ scores,
    const int* __restrict__ batch, float* __restrict__ out)
{
    const int gid = blockIdx.x;
    const int tid = threadIdx.x;
    __shared__ float red[256];
    __shared__ float s_max, s_inv;

    int lo = 0, hi = N_NODES;
    while (lo < hi) { int mid = (lo + hi) >> 1; if (batch[mid] < gid) lo = mid + 1; else hi = mid; }
    const int start = lo;
    hi = N_NODES;
    while (lo < hi) { int mid = (lo + hi) >> 1; if (batch[mid] < gid + 1) lo = mid + 1; else hi = mid; }
    const int end = lo;

    // segment max
    float lmax = -3.0e38f;
    for (int i = start + tid; i < end; i += 256) lmax = fmaxf(lmax, scores[i]);
    red[tid] = lmax; __syncthreads();
    for (int s = 128; s > 0; s >>= 1) { if (tid < s) red[tid] = fmaxf(red[tid], red[tid + s]); __syncthreads(); }
    if (tid == 0) s_max = red[0];
    __syncthreads();
    const float smax = s_max;

    // segment sum of exp
    float lsum = 0.f;
    for (int i = start + tid; i < end; i += 256) lsum += __expf(scores[i] - smax);
    __syncthreads();
    red[tid] = lsum; __syncthreads();
    for (int s = 128; s > 0; s >>= 1) { if (tid < s) red[tid] += red[tid + s]; __syncthreads(); }
    if (tid == 0) s_inv = 1.0f / (red[0] + 1e-16f);
    __syncthreads();
    const float inv = s_inv;

    // weighted pooling: out[gid, m] = sum_n softmax(n) * H[n, m]
    const int m = tid & 127, half = tid >> 7;
    float acc = 0.f;
    for (int i = start + half; i < end; i += 2)
        acc += __expf(scores[i] - smax) * inv * H[(size_t)i * M_DIM + m];
    __syncthreads();
    red[tid] = acc; __syncthreads();
    if (tid < 128) out[(size_t)gid * M_DIM + m] = red[tid] + red[tid + 128];
}

// ---------------------------------------------------------------------------
extern "C" void kernel_launch(void* const* d_in, const int* in_sizes, int n_in,
                              void* d_out, int out_size, void* d_ws, size_t ws_size,
                              hipStream_t stream) {
    const float* x     = (const float*)d_in[0];
    const float* g     = (const float*)d_in[1];
    const float* mu    = (const float*)d_in[2];
    const float* sigma = (const float*)d_in[3];
    const float* w1    = (const float*)d_in[4];
    const float* w2    = (const float*)d_in[5];
    const int*   batch = (const int*)d_in[6];
    float* out = (float*)d_out;

    unsigned char* ws = (unsigned char*)d_ws;
    float* H      = (float*)ws;                         // 200000*128*4 = 102,400,000 B
    float* scores = (float*)(ws + 102400000);           // 200000*4    =     800,000 B
    unsigned char* gT  = ws + 103200000;                // 1024*256*2  =     524,288 B
    unsigned char* w1b = ws + 103724288;                // 128*128*2   =      32,768 B

    prep_kernel<<<1088, 256, 0, stream>>>(g, w1, (unsigned short*)gT,
                                          (unsigned short*)w1b);
    // 200000 nodes / 16 per wave = 12500 waves; 8 waves/block -> 1563 blocks
    fused_node_kernel<<<1563, 256, 0, stream>>>(x, gT, mu, sigma, w1b, w2,
                                                H, scores);
    pool_kernel<<<N_GRAPHS, 256, 0, stream>>>(H, scores, batch, out);
}
